// GroupedQueryAttention_76948634075628
// MI455X (gfx1250) — compile-verified, matrix operand order checked by reading, not run on hardware
//
#include <hip/hip_runtime.h>
#include <hip/hip_bf16.h>
#include <cstdint>

#define B_    2
#define S_    2048
#define HID_  2048
#define H_    16
#define HKV_  4
#define D_    128
#define KV_   (HKV_ * D_)   // 512
#define MTOK_ (B_ * S_)     // 4096

typedef __attribute__((ext_vector_type(16))) __bf16 v16bf;
typedef __attribute__((ext_vector_type(8)))  __bf16 v8bf;
typedef __attribute__((ext_vector_type(8)))  float  v8f;

// ---------- small helpers ----------
__device__ __forceinline__ __bf16 f2bf(float f) {
  union { float f; uint32_t u; } a; a.f = f;
  uint32_t r = a.u + 0x7FFFu + ((a.u >> 16) & 1u);   // round-to-nearest-even
  uint16_t h = (uint16_t)(r >> 16);
  __bf16 o; __builtin_memcpy(&o, &h, 2); return o;
}
__device__ __forceinline__ float bf2f(__bf16 x) {
  uint16_t h; __builtin_memcpy(&h, &x, 2);
  union { uint32_t u; float f; } a; a.u = ((uint32_t)h) << 16; return a.f;
}
// Loads one WMMA bf16 fragment for this lane: two contiguous 16B chunks
// (elements 0..7 from lo, 8..15 from hi), matching the 16-bit A/B VGPR layout.
__device__ __forceinline__ v16bf ld_frag(const __bf16* lo, const __bf16* hi) {
  union { uint4 u; v8bf v; } a, b;
  a.u = *(const uint4*)lo;
  b.u = *(const uint4*)hi;
  return __builtin_shufflevector(a.v, b.v,
      0, 1, 2, 3, 4, 5, 6, 7, 8, 9, 10, 11, 12, 13, 14, 15);
}
__device__ __forceinline__ v8f wmma_bf16(v16bf a, v16bf b, v8f c) {
  return __builtin_amdgcn_wmma_f32_16x16x32_bf16(
      /*neg_a=*/false, a, /*neg_b=*/false, b,
      /*c_mod=*/(short)0, c, /*reuse_a=*/false, /*reuse_b=*/false);
}
// CDNA5 async global->LDS copy: 64B per lane (4 x b128), ASYNCcnt-tracked.
// ldsaddr is the LDS byte address (low 32 bits of the generic pointer),
// gaddr the 64-bit global address. INST_OFFSET applies to both sides.
__device__ __forceinline__ void async_copy64B(uint32_t ldsaddr, uint64_t gaddr) {
  asm volatile(
      "global_load_async_to_lds_b128 %0, %1, off\n\t"
      "global_load_async_to_lds_b128 %0, %1, off offset:16\n\t"
      "global_load_async_to_lds_b128 %0, %1, off offset:32\n\t"
      "global_load_async_to_lds_b128 %0, %1, off offset:48"
      :: "v"(ldsaddr), "v"(gaddr) : "memory");
}
__device__ __forceinline__ void wait_async0() {
  asm volatile("s_wait_asynccnt 0" ::: "memory");
}

// ---------- fp32 -> bf16 conversion ----------
__global__ void __launch_bounds__(256) cvt_f32_bf16(
    const float* __restrict__ in, __bf16* __restrict__ out, int n) {
  int i = blockIdx.x * 256 + threadIdx.x;
  int stride = gridDim.x * 256;
  for (; i < n; i += stride) out[i] = f2bf(in[i]);
}

// ---------- WMMA GEMM: C[M,N] = A[M,K] @ W[K,N] + bias ----------
// Block tile 128x64, K-slab 64, 8 waves, each wave 32x32 (2x2 WMMA tiles,
// 8 v_wmma per slab). A is staged via GLOBAL_LOAD_ASYNC_TO_LDS_B128
// (ASYNCcnt), B is staged with a transpose into [n][k] via DS stores.
template <int OUT_F32>
__global__ void __launch_bounds__(256) wmma_gemm(
    const __bf16* __restrict__ A, const __bf16* __restrict__ W,
    const float* __restrict__ bias, __bf16* __restrict__ Obf,
    float* __restrict__ Of32, int M, int N, int K)
{
  __shared__ alignas(16) __bf16 As[128][64];
  __shared__ alignas(16) __bf16 Bs[64][64];   // transposed: [n][k]
  const int t    = threadIdx.x;
  const int lane = t & 31, wave = t >> 5;
  const int ln   = lane & 15, h2 = lane >> 4;
  const int wr   = wave >> 1, wc = wave & 1;
  const int m0   = blockIdx.y * 128, n0 = blockIdx.x * 64;

  v8f acc[2][2] = { { {}, {} }, { {}, {} } };

  const int arow = t >> 1, ako = (t & 1) * 32;  // A: 2 threads/row, 32 elems (64B) each
  const int bk   = t >> 2, bng = t & 3;         // B: 16 n-elems per thread, one k row

  const uint32_t alds = (uint32_t)(uintptr_t)&As[arow][ako];
  const __bf16*  agp  = A + (size_t)(m0 + arow) * K + ako;
  const __bf16*  wgp  = W + (size_t)bk * N + n0 + bng * 16;

  for (int k0 = 0; k0 < K; k0 += 64) {
    // --- A slab: async copy straight into LDS (no VGPR staging) ---
    async_copy64B(alds, (uint64_t)(uintptr_t)(agp + k0));

    // --- B slab: coalesced b128 loads + transposing DS stores ---
    union { uint4 u; v8bf v; } bb0, bb1;
    const __bf16* wg = wgp + (size_t)k0 * N;
    bb0.u = *(const uint4*)wg;
    bb1.u = *(const uint4*)(wg + 8);
    #pragma unroll
    for (int j = 0; j < 8; ++j) {
      Bs[bng * 16 + j][bk]     = bb0.v[j];
      Bs[bng * 16 + 8 + j][bk] = bb1.v[j];
    }
    if (k0 + 64 < K) {                       // prefetch next slabs into L2/WGP$
      __builtin_prefetch(agp + k0 + 64, 0, 0);
      __builtin_prefetch(wgp + (size_t)(k0 + 64) * N, 0, 0);
    }
    wait_async0();                           // my async writes landed in LDS
    __syncthreads();                         // everyone's stores visible

    #pragma unroll
    for (int dk = 0; dk < 2; ++dk) {
      #pragma unroll
      for (int si = 0; si < 2; ++si) {
        const int mr = wr * 32 + si * 16 + ln;
        v16bf af = ld_frag(&As[mr][dk * 32 + h2 * 8],
                           &As[mr][dk * 32 + 16 + h2 * 8]);
        #pragma unroll
        for (int sj = 0; sj < 2; ++sj) {
          const int nr = wc * 32 + sj * 16 + ln;
          v16bf bf = ld_frag(&Bs[nr][dk * 32 + h2 * 8],
                             &Bs[nr][dk * 32 + 16 + h2 * 8]);
          acc[si][sj] = wmma_bf16(af, bf, acc[si][sj]);
        }
      }
    }
    __syncthreads();
  }

  #pragma unroll
  for (int si = 0; si < 2; ++si)
    #pragma unroll
    for (int sj = 0; sj < 2; ++sj)
      #pragma unroll
      for (int i = 0; i < 8; ++i) {
        int m = m0 + wr * 32 + si * 16 + i + 8 * h2;
        int n = n0 + wc * 32 + sj * 16 + ln;
        float v = acc[si][sj][i] + bias[n];
        if (OUT_F32) Of32[(size_t)m * N + n] = v;
        else         Obf[(size_t)m * N + n]  = f2bf(v);
      }
}

// ---------- RoPE + relayout [B,S,NH,D] -> [B,NH,S,D] ----------
__global__ void __launch_bounds__(256) rope_transpose(
    const __bf16* __restrict__ Lin, __bf16* __restrict__ Out, int NH, int doRope)
{
  int t = blockIdx.x * 256 + threadIdx.x;
  int d = t & 63;
  int r = t >> 6;
  int h = r % NH; r /= NH;
  int s = r % S_; int b = r / S_;
  size_t ibase = ((size_t)(b * S_ + s)) * (NH * D_) + h * D_;
  float x1 = bf2f(Lin[ibase + d]);
  float x2 = bf2f(Lin[ibase + d + 64]);
  float o1 = x1, o2 = x2;
  if (doRope) {
    float freq = (float)s * __powf(10000.0f, -(2.0f * (float)d) / 128.0f);
    float c = __cosf(freq), sn = __sinf(freq);
    o1 = x1 * c - x2 * sn;
    o2 = x2 * c + x1 * sn;
  }
  size_t obase = (((size_t)(b * NH + h)) * S_ + s) * D_ + d;
  Out[obase]      = f2bf(o1);
  Out[obase + 64] = f2bf(o2);
}

// ---------- V relayout [B,S,HKV,D] -> [B,HKV,D,S] (V^T per head) ----------
__global__ void __launch_bounds__(256) vtrans(
    const __bf16* __restrict__ Lin, __bf16* __restrict__ Out) {
  int t = blockIdx.x * 256 + threadIdx.x;        // linear in [b][h][d][s]
  int s = t & (S_ - 1);
  int d = (t >> 11) & (D_ - 1);
  int h = (t >> 18) & (HKV_ - 1);
  int b = t >> 20;
  Out[t] = Lin[((size_t)(b * S_ + s)) * KV_ + h * D_ + d];
}

// ---------- Flash attention: one wave per 16-query tile ----------
__global__ void __launch_bounds__(256) flash_attn(
    const __bf16* __restrict__ Qh, const __bf16* __restrict__ Kh,
    const __bf16* __restrict__ Vt, const float* __restrict__ mask,
    __bf16* __restrict__ O)
{
  __shared__ alignas(16) __bf16 Pt[8][16][32];   // per-wave P tile
  const int lane = threadIdx.x & 31, wave = threadIdx.x >> 5;
  const int ln = lane & 15, h2 = lane >> 4;
  const int b = blockIdx.z, h = blockIdx.y, hk = h >> 2;   // GQA group of 4
  const int q0 = blockIdx.x * 128 + wave * 16;

  const __bf16* Qrow = Qh + (((size_t)(b * H_ + h)) * S_ + q0 + ln) * D_;
  v16bf qf[4];
  #pragma unroll
  for (int dk = 0; dk < 4; ++dk)
    qf[dk] = ld_frag(Qrow + dk * 32 + h2 * 8, Qrow + dk * 32 + 16 + h2 * 8);

  v8f oacc[8] = {};
  float row_m[8], row_l[8];
  #pragma unroll
  for (int i = 0; i < 8; ++i) { row_m[i] = -3.0e38f; row_l[i] = 0.0f; }

  const float* mrow = mask + (size_t)b * S_ * S_;
  const float scl = 0.08838834764831845f;        // 1/sqrt(128)

  for (int key0 = 0; key0 < S_; key0 += 32) {
    const __bf16* Krow0 = Kh + (((size_t)(b * HKV_ + hk)) * S_ + key0 + ln) * D_;
    const __bf16* Krow1 = Krow0 + (size_t)16 * D_;
    v8f s0 = {}, s1 = {};
    #pragma unroll
    for (int dk = 0; dk < 4; ++dk) {
      v16bf k0f = ld_frag(Krow0 + dk * 32 + h2 * 8, Krow0 + dk * 32 + 16 + h2 * 8);
      s0 = wmma_bf16(qf[dk], k0f, s0);
      v16bf k1f = ld_frag(Krow1 + dk * 32 + h2 * 8, Krow1 + dk * 32 + 16 + h2 * 8);
      s1 = wmma_bf16(qf[dk], k1f, s1);
    }
    // online softmax over this 32-key block
    #pragma unroll
    for (int i = 0; i < 8; ++i) {
      int qr = q0 + i + 8 * h2;
      float v0 = s0[i] * scl + mrow[(size_t)qr * S_ + key0 + ln];
      float v1 = s1[i] * scl + mrow[(size_t)qr * S_ + key0 + 16 + ln];
      float mx = fmaxf(v0, v1);
      #pragma unroll
      for (int msk = 1; msk < 16; msk <<= 1) mx = fmaxf(mx, __shfl_xor(mx, msk, 32));
      float mn = fmaxf(row_m[i], mx);
      float alpha = __expf(row_m[i] - mn);
      row_m[i] = mn;
      float p0 = __expf(v0 - mn);
      float p1 = __expf(v1 - mn);
      float rs = p0 + p1;
      #pragma unroll
      for (int msk = 1; msk < 16; msk <<= 1) rs += __shfl_xor(rs, msk, 32);
      row_l[i] = row_l[i] * alpha + rs;
      #pragma unroll
      for (int tt = 0; tt < 8; ++tt) oacc[tt][i] *= alpha;
      Pt[wave][i + 8 * h2][ln]      = f2bf(p0);
      Pt[wave][i + 8 * h2][16 + ln] = f2bf(p1);
    }
    // P (C-layout) -> A-fragment via per-wave LDS (DS ops in-order per wave)
    v16bf pf = ld_frag(&Pt[wave][ln][h2 * 8], &Pt[wave][ln][16 + h2 * 8]);
    #pragma unroll
    for (int tt = 0; tt < 8; ++tt) {
      const __bf16* vrow =
          Vt + (((size_t)(b * HKV_ + hk)) * D_ + tt * 16 + ln) * S_ + key0;
      v16bf vf = ld_frag(vrow + h2 * 8, vrow + 16 + h2 * 8);
      oacc[tt] = wmma_bf16(pf, vf, oacc[tt]);
    }
  }
  // normalize and store as rows of [B*S, HID]
  #pragma unroll
  for (int i = 0; i < 8; ++i) {
    float inv = 1.0f / row_l[i];
    int m = q0 + i + 8 * h2;
    #pragma unroll
    for (int tt = 0; tt < 8; ++tt) {
      O[((size_t)(b * S_) + m) * HID_ + h * D_ + tt * 16 + ln] =
          f2bf(oacc[tt][i] * inv);
    }
  }
}

// ---------- host-side launcher ----------
extern "C" void kernel_launch(void* const* d_in, const int* in_sizes, int n_in,
                              void* d_out, int out_size, void* d_ws, size_t ws_size,
                              hipStream_t stream)
{
  (void)in_sizes; (void)n_in; (void)out_size; (void)ws_size;
  const float* hs   = (const float*)d_in[0];
  const float* mask = (const float*)d_in[1];
  const float* Wq   = (const float*)d_in[2];
  const float* bq   = (const float*)d_in[3];
  const float* Wk   = (const float*)d_in[4];
  const float* bk   = (const float*)d_in[5];
  const float* Wv   = (const float*)d_in[6];
  const float* bv   = (const float*)d_in[7];
  const float* Wo   = (const float*)d_in[8];
  const float* bo   = (const float*)d_in[9];

  char* w = (char*)d_ws;
  size_t off = 0;
  auto give = [&](size_t bytes) -> char* {
    char* p = w + off;
    off += (bytes + 255) & ~(size_t)255;
    return p;
  };
  __bf16* Xbf  = (__bf16*)give((size_t)MTOK_ * HID_ * 2);
  __bf16* Wqb  = (__bf16*)give((size_t)HID_ * HID_ * 2);
  __bf16* Wkb  = (__bf16*)give((size_t)HID_ * KV_ * 2);
  __bf16* Wvb  = (__bf16*)give((size_t)HID_ * KV_ * 2);
  __bf16* Wob  = (__bf16*)give((size_t)HID_ * HID_ * 2);
  __bf16* Qlin = (__bf16*)give((size_t)MTOK_ * HID_ * 2);
  __bf16* Klin = (__bf16*)give((size_t)MTOK_ * KV_ * 2);
  __bf16* Vlin = (__bf16*)give((size_t)MTOK_ * KV_ * 2);
  __bf16* Qhd  = (__bf16*)give((size_t)MTOK_ * HID_ * 2);
  __bf16* Khd  = (__bf16*)give((size_t)MTOK_ * KV_ * 2);
  __bf16* Vtd  = (__bf16*)give((size_t)MTOK_ * KV_ * 2);
  __bf16* AttnO = Xbf;   // X is dead after the QKV projections; reuse it

  cvt_f32_bf16<<<8192, 256, 0, stream>>>(hs, Xbf, MTOK_ * HID_);
  cvt_f32_bf16<<<8192, 256, 0, stream>>>(Wq, Wqb, HID_ * HID_);
  cvt_f32_bf16<<<4096, 256, 0, stream>>>(Wk, Wkb, HID_ * KV_);
  cvt_f32_bf16<<<4096, 256, 0, stream>>>(Wv, Wvb, HID_ * KV_);
  cvt_f32_bf16<<<8192, 256, 0, stream>>>(Wo, Wob, HID_ * HID_);

  wmma_gemm<0><<<dim3(HID_ / 64, MTOK_ / 128), 256, 0, stream>>>(
      Xbf, Wqb, bq, Qlin, nullptr, MTOK_, HID_, HID_);
  wmma_gemm<0><<<dim3(KV_ / 64, MTOK_ / 128), 256, 0, stream>>>(
      Xbf, Wkb, bk, Klin, nullptr, MTOK_, KV_, HID_);
  wmma_gemm<0><<<dim3(KV_ / 64, MTOK_ / 128), 256, 0, stream>>>(
      Xbf, Wvb, bv, Vlin, nullptr, MTOK_, KV_, HID_);

  rope_transpose<<<(B_ * S_ * H_ * 64) / 256, 256, 0, stream>>>(Qlin, Qhd, H_, 1);
  rope_transpose<<<(B_ * S_ * HKV_ * 64) / 256, 256, 0, stream>>>(Klin, Khd, HKV_, 1);
  vtrans<<<(B_ * S_ * KV_) / 256, 256, 0, stream>>>(Vlin, Vtd);

  flash_attn<<<dim3(S_ / 128, H_, B_), 256, 0, stream>>>(Qhd, Khd, Vtd, mask, AttnO);

  wmma_gemm<1><<<dim3(HID_ / 64, MTOK_ / 128), 256, 0, stream>>>(
      AttnO, Wob, bo, nullptr, (float*)d_out, MTOK_, HID_, HID_);
}
